// recoAnomaly_38208029065461
// MI455X (gfx1250) — matrix operand order repaired and checked by reading, not compiled, not run on hardware
//
#include <hip/hip_runtime.h>

typedef __attribute__((ext_vector_type(16))) __bf16 v16bf;
typedef __attribute__((ext_vector_type(8)))  float  v8f;

#define N_NODES 50000
#define IN_C 128
#define L1C 96
#define OUT_C 64
#define EPS_F 1e-5f

// ---------------- small utility kernels ----------------

__global__ void k_zero(float* __restrict__ p, int n) {
  int i = blockIdx.x * blockDim.x + threadIdx.x;
  if (i < n) p[i] = 0.0f;
}

__global__ void k_deg_init(float* __restrict__ deg, int n) {
  int i = blockIdx.x * blockDim.x + threadIdx.x;
  if (i < n) deg[i] = 1.0f;  // self loop contributes 1 to in-degree
}

__global__ void k_deg_edges(const int* __restrict__ dst, float* __restrict__ deg, int E) {
  int e = blockIdx.x * blockDim.x + threadIdx.x;
  if (e < E) atomicAdd(&deg[dst[e]], 1.0f);
}

__global__ void k_dinv(const float* __restrict__ deg, float* __restrict__ dinv, int n) {
  int i = blockIdx.x * blockDim.x + threadIdx.x;
  if (i < n) dinv[i] = rsqrtf(deg[i]);  // deg >= 1 always
}

// ---------------- BatchNorm statistics + fold into W1 ----------------

__global__ void k_bn_reduce(const float* __restrict__ x,
                            float* __restrict__ bnsum, float* __restrict__ bnsq) {
  __shared__ float sh[512];
  int tid = threadIdx.x;       // 256 threads: 128 channels x 2 row-parts
  int c = tid & 127;
  int rpart = tid >> 7;
  float s = 0.0f, ss = 0.0f;
  for (int r = blockIdx.x * 2 + rpart; r < N_NODES; r += gridDim.x * 2) {
    float v = x[(size_t)r * IN_C + c];
    s += v; ss += v * v;
  }
  sh[tid] = s; sh[256 + tid] = ss;
  __syncthreads();
  if (tid < 128) {
    atomicAdd(&bnsum[c], sh[tid] + sh[tid + 128]);
    atomicAdd(&bnsq[c],  sh[256 + tid] + sh[256 + tid + 128]);
  }
}

// one block of 128 threads: BN(x)@W1 = x @ (scale*W1) + 1 (x) t1
__global__ void k_bn_fold(const float* __restrict__ bnsum, const float* __restrict__ bnsq,
                          const float* __restrict__ bnw, const float* __restrict__ bnb,
                          const float* __restrict__ W1,
                          __bf16* __restrict__ W1s, float* __restrict__ t1) {
  __shared__ float scale[IN_C], shift[IN_C];
  int tid = threadIdx.x;
  if (tid < IN_C) {
    float inv_n = 1.0f / (float)N_NODES;
    float mean = bnsum[tid] * inv_n;
    float var  = bnsq[tid] * inv_n - mean * mean;
    float sc = rsqrtf(var + EPS_F) * bnw[tid];
    scale[tid] = sc;
    shift[tid] = bnb[tid] - mean * sc;
  }
  __syncthreads();
  for (int idx = tid; idx < IN_C * L1C; idx += blockDim.x)
    W1s[idx] = (__bf16)(scale[idx / L1C] * W1[idx]);
  for (int j = tid; j < L1C; j += blockDim.x) {
    float acc = 0.0f;
    for (int c = 0; c < IN_C; ++c) acc += shift[c] * W1[c * L1C + j];
    t1[j] = acc;
  }
}

// one block of 128 threads: graph-LN fold into W2
__global__ void k_ln_fold(const float* __restrict__ lnstats,  // [0]=sum, [1]=sumsq
                          const float* __restrict__ lnw, const float* __restrict__ lnb,
                          const float* __restrict__ W2,
                          __bf16* __restrict__ W2s, float* __restrict__ t2) {
  __shared__ float srow[L1C], crow[L1C];
  int tid = threadIdx.x;
  float inv_n = 1.0f / ((float)N_NODES * (float)L1C);
  float mean = lnstats[0] * inv_n;
  float var  = lnstats[1] * inv_n - mean * mean;
  float inv = rsqrtf(var + EPS_F);
  if (tid < L1C) {
    float sr = inv * lnw[tid];
    srow[tid] = sr;
    crow[tid] = lnb[tid] - mean * sr;
  }
  __syncthreads();
  for (int idx = tid; idx < L1C * OUT_C; idx += blockDim.x)
    W2s[idx] = (__bf16)(srow[idx / OUT_C] * W2[idx]);
  for (int k = tid; k < OUT_C; k += blockDim.x) {
    float acc = 0.0f;
    for (int j = 0; j < L1C; ++j) acc += crow[j] * W2[j * OUT_C + k];
    t2[k] = acc;
  }
}

// ---------------- bf16 WMMA GEMM: D[M x COUT] = bf16(A[M x CIN]) @ W + 1(x)t ----
// One wave per 16-row strip; wave loads A fragments once, iterates all N tiles.
// A fragment layout (16-bit 16x32): lane<16 -> K 0..7 / 16..23, lane>=16 -> 8..15 / 24..31
// B fragment layout (16-bit 32x16): lane%16 = N, lane<16 -> K 0..7 / 16..23, lane>=16 -> 8..15 / 24..31
template <int CIN, int COUT>
__global__ __launch_bounds__(128) void k_gemm(const float* __restrict__ A,
                                              const __bf16* __restrict__ Bw,
                                              const float* __restrict__ t,
                                              float* __restrict__ D) {
  constexpr int KC = CIN / 32;
  constexpr int NT = COUT / 16;
  int wave = threadIdx.x >> 5;
  int lane = threadIdx.x & 31;
  int strip = blockIdx.x * 4 + wave;
  if (strip >= N_NODES / 16) return;   // wave-uniform: EXEC stays all-ones for WMMA
  int hi  = lane >> 4;
  int col = lane & 15;
  const float* ar = A + (size_t)(strip * 16 + col) * CIN;

  v16bf afrag[KC];
#pragma unroll
  for (int kc = 0; kc < KC; ++kc) {
    int kb = kc * 32 + hi * 8;
#pragma unroll
    for (int i = 0; i < 8; ++i) afrag[kc][i]     = (__bf16)ar[kb + i];
#pragma unroll
    for (int i = 0; i < 8; ++i) afrag[kc][8 + i] = (__bf16)ar[kb + 16 + i];
  }

#pragma unroll
  for (int nt = 0; nt < NT; ++nt) {
    v8f acc = {};
#pragma unroll
    for (int kc = 0; kc < KC; ++kc) {
      v16bf bfrag;
#pragma unroll
      for (int i = 0; i < 8; ++i) {
        bfrag[i]     = Bw[(size_t)(kc * 32 + hi * 8 + i) * COUT + nt * 16 + col];
        bfrag[8 + i] = Bw[(size_t)(kc * 32 + hi * 8 + 16 + i) * COUT + nt * 16 + col];
      }
      acc = __builtin_amdgcn_wmma_f32_16x16x32_bf16(
          false, afrag[kc], false, bfrag, (short)0, acc, false, false);
    }
    float tb = t[nt * 16 + col];
#pragma unroll
    for (int r = 0; r < 8; ++r) {
      int grow = strip * 16 + hi * 8 + r;
      D[(size_t)grow * COUT + nt * 16 + col] = acc[r] + tb;
    }
  }
}

// ---------------- aggregation ----------------

template <int NCH>
__global__ void k_selfloop(const float* __restrict__ h, const float* __restrict__ dinv,
                           float* __restrict__ agg, int total) {
  int i = blockIdx.x * blockDim.x + threadIdx.x;
  if (i < total) {
    float d = dinv[i / NCH];
    agg[i] = d * d * h[i];
  }
}

template <int NCH>
__global__ void k_edges(const int* __restrict__ src, const int* __restrict__ dst,
                        const float* __restrict__ dinv,
                        const float* __restrict__ h, float* __restrict__ agg, int E) {
  constexpr int CH = NCH / 4;
  int idx = blockIdx.x * blockDim.x + threadIdx.x;
  int e = idx / CH;
  if (e >= E) return;
  int j = (idx - e * CH) * 4;
  int s = src[e], d = dst[e];
  float w = dinv[s] * dinv[d];
  const float4 v = *(const float4*)(h + (size_t)s * NCH + j);
  float* o = agg + (size_t)d * NCH + j;
  atomicAdd(o + 0, w * v.x);
  atomicAdd(o + 1, w * v.y);
  atomicAdd(o + 2, w * v.z);
  atomicAdd(o + 3, w * v.w);
}

// bias + ReLU in place, plus scalar LN statistics for next layer
__global__ void k_post1(float* __restrict__ h, const float* __restrict__ b1,
                        float* __restrict__ lnstats, int total) {
  __shared__ float sh[512];
  int tid = threadIdx.x;
  float s = 0.0f, ss = 0.0f;
  for (int i = blockIdx.x * blockDim.x + tid; i < total; i += gridDim.x * blockDim.x) {
    float v = h[i] + b1[i % L1C];
    v = v > 0.0f ? v : 0.0f;
    h[i] = v;
    s += v; ss += v * v;
  }
  sh[tid] = s; sh[256 + tid] = ss;
  __syncthreads();
  for (int off = 128; off > 0; off >>= 1) {
    if (tid < off) { sh[tid] += sh[tid + off]; sh[256 + tid] += sh[256 + tid + off]; }
    __syncthreads();
  }
  if (tid == 0) { atomicAdd(&lnstats[0], sh[0]); atomicAdd(&lnstats[1], sh[256]); }
}

__global__ void k_out(float* __restrict__ out, const float* __restrict__ b2, int total) {
  int i = blockIdx.x * blockDim.x + threadIdx.x;
  if (i < total) {
    float v = out[i] + b2[i & (OUT_C - 1)];
    out[i] = v > 0.0f ? v : 0.0f;
  }
}

// ---------------- launch ----------------

extern "C" void kernel_launch(void* const* d_in, const int* in_sizes, int n_in,
                              void* d_out, int out_size, void* d_ws, size_t ws_size,
                              hipStream_t stream) {
  const float* x   = (const float*)d_in[0];
  const int*   ei  = (const int*)d_in[1];
  const float* bnw = (const float*)d_in[2];
  const float* bnb = (const float*)d_in[3];
  const float* W1  = (const float*)d_in[4];
  const float* b1  = (const float*)d_in[5];
  const float* lnw = (const float*)d_in[6];
  const float* lnb = (const float*)d_in[7];
  const float* W2  = (const float*)d_in[8];
  const float* b2  = (const float*)d_in[9];
  float* out = (float*)d_out;

  const int E = in_sizes[1] / 2;
  const int* src = ei;
  const int* dst = ei + E;

  // workspace carving (256B aligned regions)
  char* base = (char*)d_ws;
  size_t off = 0;
  auto alloc = [&](size_t bytes) -> void* {
    void* p = base + off;
    off = (off + bytes + 255) & ~(size_t)255;
    return p;
  };
  float*  deg   = (float*)alloc((size_t)N_NODES * 4);
  float*  dinv  = (float*)alloc((size_t)N_NODES * 4);
  float*  stats = (float*)alloc(272 * 4);      // [0..127]=bnsum [128..255]=bnsq [256..257]=ln
  float*  t1    = (float*)alloc(L1C * 4);
  float*  t2    = (float*)alloc(OUT_C * 4);
  __bf16* W1s   = (__bf16*)alloc((size_t)IN_C * L1C * 2);
  __bf16* W2s   = (__bf16*)alloc((size_t)L1C * OUT_C * 2);
  float*  h1    = (float*)alloc((size_t)N_NODES * L1C * 4);   // pre-aggregation layer1
  float*  agg1  = (float*)alloc((size_t)N_NODES * L1C * 4);   // becomes h2 (bias+relu in place)
  float*  h3    = (float*)alloc((size_t)N_NODES * OUT_C * 4); // pre-aggregation layer2

  const int T1 = N_NODES * L1C;     // 4.8M
  const int T2 = N_NODES * OUT_C;   // 3.2M

  // degrees + dinv
  k_deg_init<<<(N_NODES + 255) / 256, 256, 0, stream>>>(deg, N_NODES);
  k_zero<<<2, 256, 0, stream>>>(stats, 258);
  k_deg_edges<<<(E + 255) / 256, 256, 0, stream>>>(dst, deg, E);
  k_dinv<<<(N_NODES + 255) / 256, 256, 0, stream>>>(deg, dinv, N_NODES);

  // BN stats + fold into W1
  k_bn_reduce<<<256, 256, 0, stream>>>(x, stats, stats + 128);
  k_bn_fold<<<1, 128, 0, stream>>>(stats, stats + 128, bnw, bnb, W1, W1s, t1);

  // layer 1: GEMM (WMMA bf16) -> self-loop init -> edge scatter -> bias+relu+LN stats
  k_gemm<IN_C, L1C><<<(N_NODES / 16 + 3) / 4, 128, 0, stream>>>(x, W1s, t1, h1);
  k_selfloop<L1C><<<(T1 + 255) / 256, 256, 0, stream>>>(h1, dinv, agg1, T1);
  k_edges<L1C><<<(E * (L1C / 4) + 255) / 256, 256, 0, stream>>>(src, dst, dinv, h1, agg1, E);
  k_post1<<<512, 256, 0, stream>>>(agg1, b1, stats + 256, T1);

  // LN fold into W2
  k_ln_fold<<<1, 128, 0, stream>>>(stats + 256, lnw, lnb, W2, W2s, t2);

  // layer 2: GEMM (WMMA bf16) -> self-loop init (writes d_out) -> scatter -> bias+relu
  k_gemm<L1C, OUT_C><<<(N_NODES / 16 + 3) / 4, 128, 0, stream>>>(agg1, W2s, t2, h3);
  k_selfloop<OUT_C><<<(T2 + 255) / 256, 256, 0, stream>>>(h3, dinv, out, T2);
  k_edges<OUT_C><<<(E * (OUT_C / 4) + 255) / 256, 256, 0, stream>>>(src, dst, dinv, h3, out, E);
  k_out<<<(T2 + 255) / 256, 256, 0, stream>>>(out, b2, T2);
}
